// GPRGNN_4501125726324
// MI455X (gfx1250) — matrix operand order repaired and checked by reading, not compiled, not run on hardware
//
#include <hip/hip_runtime.h>
#include <hip/hip_bf16.h>

typedef __bf16 bf16_t;
typedef __attribute__((ext_vector_type(16))) __bf16 v16bf;
typedef __attribute__((ext_vector_type(8)))  float  v8f;

#define IN_C  512
#define HID_C 256
#define OUT_C 64
#define KHOP  10

#define LDS_PITCH 264   // 256 + 8 bf16 pad -> 132 dwords/row, 132%64==4: conflict-free b128 frag reads
#define USE_ASYNC_LDS 1

#define WMMA_BF16(A, B, C) \
    __builtin_amdgcn_wmma_f32_16x16x32_bf16(false, (A), false, (B), (short)0, (C), false, false)

// ----------------------------- conversions ---------------------------------
__global__ void cvt_f32_bf16(const float* __restrict__ src, bf16_t* __restrict__ dst, int n) {
    int i = blockIdx.x * blockDim.x + threadIdx.x;
    if (i < n) dst[i] = (bf16_t)src[i];
}

// ----------------------------- gcn norm ------------------------------------
__global__ void deg_init(float* __restrict__ deg, int n) {
    int i = blockIdx.x * blockDim.x + threadIdx.x;
    if (i < n) deg[i] = 1.0f;                      // self-loop
}
__global__ void deg_edges(const long long* __restrict__ dst, float* __restrict__ deg, int e) {
    int i = blockIdx.x * blockDim.x + threadIdx.x;
    if (i < e) unsafeAtomicAdd(&deg[(int)dst[i]], 1.0f);
}
__global__ void deg_rsqrt(float* __restrict__ deg, int n) {
    int i = blockIdx.x * blockDim.x + threadIdx.x;
    if (i < n) deg[i] = rsqrtf(deg[i]);            // deg >= 1 always (self-loop)
}

// --------------------------- fragment loaders ------------------------------
// A: 16x32 bf16 (MxK). laneBase = X + row*lda + (lane&16 ? 16 : 0).
// lanes 0-15: K=k..k+15; lanes 16-31: K=k+16..k+31
__device__ __forceinline__ v16bf load_frag_a(const bf16_t* __restrict__ laneBase, int k) {
    const bf16_t* p = laneBase + k;
    union { v16bf v; uint4 u[2]; } t;
    t.u[0] = *(const uint4*)(p);
    t.u[1] = *(const uint4*)(p + 8);
    return t.v;
}
// B: 32x16 bf16 (KxN) from LDS tile sW[col][LDS_PITCH]; col = tile n0 + (lane&15),
// k = kk + (lane&16 ? 16 : 0): 32 contiguous bytes per lane, 2x ds_load_b128.
__device__ __forceinline__ v16bf load_frag_b_lds(const bf16_t* s, int col, int k) {
    const bf16_t* p = s + col * LDS_PITCH + k;
    union { v16bf v; uint4 u[2]; } t;
    t.u[0] = *(const uint4*)(p);
    t.u[1] = *(const uint4*)(p + 8);
    return t.v;
}

// cooperative LDS fill of one 64-col x 256-K weight slice (CDNA5 async path)
__device__ __forceinline__ void fill_w_slice(const bf16_t* __restrict__ W, int ldw,
                                             int nbase, int kc, bf16_t* s, int tid) {
    int c = tid >> 2;              // 0..63 local column
    int q = (tid & 3) * 64;        // K quarter within the 256-wide chunk
    const bf16_t* g = W + (long)(nbase + c) * ldw + kc + q;
    bf16_t* d = s + c * LDS_PITCH + q;
    #pragma unroll
    for (int j = 0; j < 64; j += 8) {
#if USE_ASYNC_LDS
        unsigned int  ldsOff = (unsigned int)(unsigned long long)(d + j);
        unsigned long long ga = (unsigned long long)(g + j);
        asm volatile("global_load_async_to_lds_b128 %0, %1, off"
                     :: "v"(ldsOff), "v"(ga) : "memory");
#else
        *(uint4*)(d + j) = *(const uint4*)(g + j);
#endif
    }
}
__device__ __forceinline__ void wait_fill() {
#if USE_ASYNC_LDS
    asm volatile("s_wait_asynccnt 0x0" ::: "memory");
#endif
    __syncthreads();
}

// --------------------------- GEMM1: h1 = relu(x @ W1^T + b1) ----------------
// grid = mgroups*4 (ngroup fastest -> A slabs L2-reused across the 4 col groups)
// block = 8 waves; wave w -> mslab = mgroup*8 + w; covers 16 rows x 64 cols.
__global__ __launch_bounds__(256)
void gemm1_wmma(const bf16_t* __restrict__ X, const bf16_t* __restrict__ W1,
                const float* __restrict__ b1, bf16_t* __restrict__ H1, int M) {
    __shared__ __align__(16) bf16_t sW[64 * LDS_PITCH];
    int tid  = threadIdx.x;
    int lane = tid & 31;
    int wave = tid >> 5;
    int ngroup = blockIdx.x & 3;
    int mgroup = blockIdx.x >> 2;
    int mslab  = mgroup * 8 + wave;
    int m0     = mslab * 16;
    int nbase  = ngroup * 64;
    bool active = (m0 < M);                        // wave-uniform

    const bf16_t* aBase = X + (long)(m0 + (lane & 15)) * IN_C + ((lane & 16) ? 16 : 0);
    int bcol = lane & 15;
    int bk   = (lane & 16) ? 16 : 0;

    v8f acc0 = {}, acc1 = {}, acc2 = {}, acc3 = {};

    for (int kc = 0; kc < IN_C; kc += 256) {
        __syncthreads();                            // previous chunk fully consumed
        fill_w_slice(W1, IN_C, nbase, kc, sW, tid);
        wait_fill();
        if (active) {
            // software pipeline: A and all 4 B fragments double-buffered in regs
            v16bf aCur = load_frag_a(aBase, kc);
            v16bf c0 = load_frag_b_lds(sW,  0 + bcol, bk);
            v16bf c1 = load_frag_b_lds(sW, 16 + bcol, bk);
            v16bf c2 = load_frag_b_lds(sW, 32 + bcol, bk);
            v16bf c3 = load_frag_b_lds(sW, 48 + bcol, bk);
            #pragma unroll
            for (int kk = 0; kk < 256; kk += 32) {
                v16bf a  = aCur;
                v16bf b0 = c0, b1f = c1, b2f = c2, b3f = c3;
                if (kk + 32 < 256) {
                    aCur = load_frag_a(aBase, kc + kk + 32);
                    c0 = load_frag_b_lds(sW,  0 + bcol, kk + 32 + bk);
                    c1 = load_frag_b_lds(sW, 16 + bcol, kk + 32 + bk);
                    c2 = load_frag_b_lds(sW, 32 + bcol, kk + 32 + bk);
                    c3 = load_frag_b_lds(sW, 48 + bcol, kk + 32 + bk);
                }
                acc0 = WMMA_BF16(a, b0,  acc0);
                acc1 = WMMA_BF16(a, b1f, acc1);
                acc2 = WMMA_BF16(a, b2f, acc2);
                acc3 = WMMA_BF16(a, b3f, acc3);
            }
        }
    }
    if (!active) return;

    // C/D layout: VGPR r, lanes 0-15 -> M=r, N=lane; lanes 16-31 -> M=8+r, N=lane-16
    int col = lane & 15;
    int rowBase = m0 + ((lane & 16) ? 8 : 0);
    #pragma unroll
    for (int r = 0; r < 8; ++r) {
        long row = rowBase + r;
        float v0 = acc0[r] + b1[nbase +  0 + col]; v0 = v0 > 0.f ? v0 : 0.f;
        float v1 = acc1[r] + b1[nbase + 16 + col]; v1 = v1 > 0.f ? v1 : 0.f;
        float v2 = acc2[r] + b1[nbase + 32 + col]; v2 = v2 > 0.f ? v2 : 0.f;
        float v3 = acc3[r] + b1[nbase + 48 + col]; v3 = v3 > 0.f ? v3 : 0.f;
        H1[row * HID_C + nbase +  0 + col] = (bf16_t)v0;
        H1[row * HID_C + nbase + 16 + col] = (bf16_t)v1;
        H1[row * HID_C + nbase + 32 + col] = (bf16_t)v2;
        H1[row * HID_C + nbase + 48 + col] = (bf16_t)v3;
    }
}

// ------------- GEMM2: h = h1 @ W2^T + b2 ; xk = h ; hidden = temp[0]*h ------
// W2 (64x256) fits in a single LDS slice. block = 8 waves x 8 mslabs, all 64 cols.
__global__ __launch_bounds__(256)
void gemm2_wmma(const bf16_t* __restrict__ H1, const bf16_t* __restrict__ W2,
                const float* __restrict__ b2, const float* __restrict__ temp,
                float* __restrict__ Xk, float* __restrict__ Hid, int M) {
    __shared__ __align__(16) bf16_t sW[64 * LDS_PITCH];
    int tid  = threadIdx.x;
    int lane = tid & 31;
    int wave = tid >> 5;
    int mslab = blockIdx.x * 8 + wave;
    int m0 = mslab * 16;
    bool active = (m0 < M);                        // wave-uniform

    fill_w_slice(W2, HID_C, 0, 0, sW, tid);
    wait_fill();

    const bf16_t* aBase = H1 + (long)(m0 + (lane & 15)) * HID_C + ((lane & 16) ? 16 : 0);
    int bcol = lane & 15;
    int bk   = (lane & 16) ? 16 : 0;

    v8f acc0 = {}, acc1 = {}, acc2 = {}, acc3 = {};
    if (active) {
        v16bf aCur = load_frag_a(aBase, 0);
        v16bf c0 = load_frag_b_lds(sW,  0 + bcol, bk);
        v16bf c1 = load_frag_b_lds(sW, 16 + bcol, bk);
        v16bf c2 = load_frag_b_lds(sW, 32 + bcol, bk);
        v16bf c3 = load_frag_b_lds(sW, 48 + bcol, bk);
        #pragma unroll
        for (int kk = 0; kk < HID_C; kk += 32) {
            v16bf a  = aCur;
            v16bf b0 = c0, b1f = c1, b2f = c2, b3f = c3;
            if (kk + 32 < HID_C) {
                aCur = load_frag_a(aBase, kk + 32);
                c0 = load_frag_b_lds(sW,  0 + bcol, kk + 32 + bk);
                c1 = load_frag_b_lds(sW, 16 + bcol, kk + 32 + bk);
                c2 = load_frag_b_lds(sW, 32 + bcol, kk + 32 + bk);
                c3 = load_frag_b_lds(sW, 48 + bcol, kk + 32 + bk);
            }
            acc0 = WMMA_BF16(a, b0,  acc0);
            acc1 = WMMA_BF16(a, b1f, acc1);
            acc2 = WMMA_BF16(a, b2f, acc2);
            acc3 = WMMA_BF16(a, b3f, acc3);
        }
    }
    if (!active) return;

    float t0 = temp[0];
    int col = lane & 15;
    int rowBase = m0 + ((lane & 16) ? 8 : 0);
    #pragma unroll
    for (int r = 0; r < 8; ++r) {
        long row = rowBase + r;
        float v0 = acc0[r] + b2[ 0 + col];
        float v1 = acc1[r] + b2[16 + col];
        float v2 = acc2[r] + b2[32 + col];
        float v3 = acc3[r] + b2[48 + col];
        long o = row * OUT_C + col;
        Xk[o +  0] = v0;  Hid[o +  0] = t0 * v0;
        Xk[o + 16] = v1;  Hid[o + 16] = t0 * v1;
        Xk[o + 32] = v2;  Hid[o + 32] = t0 * v2;
        Xk[o + 48] = v3;  Hid[o + 48] = t0 * v3;
    }
}

// --------------------------- propagation -----------------------------------
// xnext = dinv[i]^2 * xk  (self-loop term; also serves as the zero init)
__global__ void prop_selfloop(const float* __restrict__ xk, const float* __restrict__ dinv,
                              float* __restrict__ xnext, int nf4) {
    int i = blockIdx.x * blockDim.x + threadIdx.x;   // nf4 = N*OUT_C/4 float4's
    if (i >= nf4) return;
    int node = i >> 4;                               // 16 float4 per node
    float w = dinv[node]; w = w * w;
    float4 v = ((const float4*)xk)[i];
    v.x *= w; v.y *= w; v.z *= w; v.w *= w;
    ((float4*)xnext)[i] = v;
}

// xnext[dst] += dinv[src]*dinv[dst] * xk[src]  — 16 threads/edge, float4 each
__global__ void prop_scatter(const long long* __restrict__ src, const long long* __restrict__ dst,
                             const float* __restrict__ dinv, const float* __restrict__ xk,
                             float* __restrict__ xnext, int e) {
    int gid = blockIdx.x * blockDim.x + threadIdx.x;
    int edge = gid >> 4;
    if (edge >= e) return;
    int c4 = gid & 15;
    int s = (int)src[edge];
    int d = (int)dst[edge];
    float w = dinv[s] * dinv[d];
    float4 v = ((const float4*)(xk + (long)s * OUT_C))[c4];
    float* p = xnext + (long)d * OUT_C + c4 * 4;
    unsafeAtomicAdd(p + 0, w * v.x);
    unsafeAtomicAdd(p + 1, w * v.y);
    unsafeAtomicAdd(p + 2, w * v.z);
    unsafeAtomicAdd(p + 3, w * v.w);
}

// hidden += temp[k] * xnext
__global__ void prop_axpy(float* __restrict__ hid, const float* __restrict__ xnext,
                          const float* __restrict__ temp, int k, int nf4) {
    int i = blockIdx.x * blockDim.x + threadIdx.x;
    if (i >= nf4) return;
    float g = temp[k];
    float4 h = ((const float4*)hid)[i];
    float4 x = ((const float4*)xnext)[i];
    h.x += g * x.x; h.y += g * x.y; h.z += g * x.z; h.w += g * x.w;
    ((float4*)hid)[i] = h;
}

// ---------------------------------------------------------------------------
extern "C" void kernel_launch(void* const* d_in, const int* in_sizes, int n_in,
                              void* d_out, int out_size, void* d_ws, size_t ws_size,
                              hipStream_t stream) {
    const float*     x    = (const float*)d_in[0];
    const long long* ei   = (const long long*)d_in[1];
    const float*     W1   = (const float*)d_in[2];
    const float*     b1   = (const float*)d_in[3];
    const float*     W2   = (const float*)d_in[4];
    const float*     b2   = (const float*)d_in[5];
    const float*     temp = (const float*)d_in[6];

    const int N = in_sizes[0] / IN_C;     // 100000
    const int E = in_sizes[1] / 2;        // 1600000
    const long long* srcI = ei;
    const long long* dstI = ei + E;
    float* hid = (float*)d_out;

    // workspace carve-out (256B aligned)
    char* ws = (char*)d_ws;
    size_t off = 0;
    auto carve = [&](size_t bytes) -> char* {
        char* p = ws + off;
        off += (bytes + 255) & ~(size_t)255;
        return p;
    };
    bf16_t* xb   = (bf16_t*)carve((size_t)N * IN_C  * sizeof(bf16_t));
    bf16_t* w1b  = (bf16_t*)carve((size_t)HID_C * IN_C  * sizeof(bf16_t));
    bf16_t* w2b  = (bf16_t*)carve((size_t)OUT_C * HID_C * sizeof(bf16_t));
    bf16_t* h1b  = (bf16_t*)carve((size_t)N * HID_C * sizeof(bf16_t));
    float*  dinv = (float*) carve((size_t)N * sizeof(float));       // deg -> dinv in place
    float*  xk   = (float*) carve((size_t)N * OUT_C * sizeof(float));
    float*  xn   = (float*) carve((size_t)N * OUT_C * sizeof(float));

    const int T = 256;
    // f32 -> bf16
    cvt_f32_bf16<<<(int)(((size_t)N * IN_C + T - 1) / T), T, 0, stream>>>(x,  xb,  N * IN_C);
    cvt_f32_bf16<<<(HID_C * IN_C  + T - 1) / T, T, 0, stream>>>(W1, w1b, HID_C * IN_C);
    cvt_f32_bf16<<<(OUT_C * HID_C + T - 1) / T, T, 0, stream>>>(W2, w2b, OUT_C * HID_C);

    // degree / norm
    deg_init <<<(N + T - 1) / T, T, 0, stream>>>(dinv, N);
    deg_edges<<<(E + T - 1) / T, T, 0, stream>>>(dstI, dinv, E);
    deg_rsqrt<<<(N + T - 1) / T, T, 0, stream>>>(dinv, N);

    // MLP via WMMA
    int mslabs  = (N + 15) / 16;                       // 6250
    int mgroups = (mslabs + 7) / 8;                    // 782
    gemm1_wmma<<<mgroups * 4, T, 0, stream>>>(xb, w1b, b1, h1b, N);
    gemm2_wmma<<<mgroups,     T, 0, stream>>>(h1b, w2b, b2, temp, xk, hid, N);

    // propagation: hidden += temp[k] * (Â^k h)
    int nf4 = N * OUT_C / 4;
    for (int k = 1; k <= KHOP; ++k) {
        prop_selfloop<<<(nf4 + T - 1) / T, T, 0, stream>>>(xk, dinv, xn, nf4);
        prop_scatter <<<(int)(((size_t)E * 16 + T - 1) / T), T, 0, stream>>>(srcI, dstI, dinv, xk, xn, E);
        prop_axpy    <<<(nf4 + T - 1) / T, T, 0, stream>>>(hid, xn, temp, k, nf4);
        float* t = xk; xk = xn; xn = t;
    }
}
